// DFA2_33414845563412
// MI455X (gfx1250) — matrix operand order repaired and checked by reading, not compiled, not run on hardware
//
#include <hip/hip_runtime.h>

// ---------------------------------------------------------------------------
// DFA chain q <- delta[sym] @ q, restructured as parallel segment products:
//   M_p = delta[s_{p,L-1}] @ ... @ delta[s_{p,0}]   (P independent WMMA GEMM chains)
//   q   = M_{P-1} @ ... @ M_0 @ e0 ;  out = dot(q, f)
// f16 operands / f32 accumulation via V_WMMA_F32_16X16X32_F16 (wave32).
// Panel staging uses GLOBAL_LOAD_ASYNC_TO_LDS_B128 + ASYNCcnt double buffering.
// LDS panels are K-chunk-major so every inner-loop ds_load is base+immediate.
// ---------------------------------------------------------------------------

typedef _Float16 half_t;
typedef __attribute__((ext_vector_type(8)))  _Float16 v8h;
typedef __attribute__((ext_vector_type(16))) _Float16 v16h;
typedef __attribute__((ext_vector_type(8)))  float    v8f;

// The async-LDS builtins take pointers to 4xi32 vectors (per hipcc diagnostic).
typedef int async_v4i __attribute__((vector_size(4 * sizeof(int))));
#define AS1 __attribute__((address_space(1)))
#define AS3 __attribute__((address_space(3)))

#define NS       512
#define NSYMS    64
#define SEQLEN   8192
#define PANEL    128
#define NPANEL   (NS / PANEL)          // 4
#define TPB      512                   // 16 waves (wave32)
#define COL_STR  40                    // 32 halfs + 8 pad: bank-conflict-free column slice
#define CHUNK_H  (PANEL * COL_STR)     // halfs per 32-K chunk  (5120)
#define PANEL_H  (16 * CHUNK_H)        // halfs per LDS panel buffer (81920 = 160 KB)
// LDS offset (halfs) of element (column c, k):  (k>>5)*CHUNK_H + c*COL_STR + (k&31)

#if defined(__HIP_DEVICE_COMPILE__) && \
    __has_builtin(__builtin_amdgcn_global_load_async_to_lds_b128) && \
    __has_builtin(__builtin_amdgcn_s_wait_asynccnt)
#define USE_ASYNC 1
#else
#define USE_ASYNC 0
#endif

#define WMMA_F16(a, b, c) \
    __builtin_amdgcn_wmma_f32_16x16x32_f16(false, (a), false, (b), (short)0, (c), false, false)

#define CAT16(lo8, hi8) \
    __builtin_shufflevector((lo8), (hi8), 0,1,2,3,4,5,6,7,8,9,10,11,12,13,14,15)

// ---------------------------------------------------------------- convert f32->f16
__global__ void dfa_convert_kernel(const float* __restrict__ src,
                                   half_t* __restrict__ dst, int n) {
    int i = blockIdx.x * blockDim.x + threadIdx.x;
    int stride = gridDim.x * blockDim.x;
    for (; i < n; i += stride) dst[i] = (half_t)src[i];
}

// ------------------------------------------- init: M_p(0) = delta[s_{p,0}]^T (col-major)
__global__ void dfa_init_kernel(const int* __restrict__ syms,
                                const half_t* __restrict__ Af16,
                                half_t* __restrict__ bufA, int lseg) {
    const int p  = blockIdx.x;
    const int s0 = syms[p * lseg];
    const half_t* A = Af16 + (size_t)s0 * NS * NS;   // row-major
    half_t* M = bufA + (size_t)p * NS * NS;          // col-major: M[c*NS + m]
    for (int e = threadIdx.x; e < NS * NS; e += blockDim.x) {
        int c = e >> 9;
        int m = e & (NS - 1);
        M[e] = A[(size_t)m * NS + c];
    }
}

// -------------------------------------------------- async panel staging (ASYNCcnt)
__device__ __forceinline__ void stage_panel_async(const half_t* __restrict__ src,
                                                  int np, half_t* lbuf, int tid) {
#if USE_ASYNC
    #pragma unroll
    for (int i = 0; i < (PANEL * NS) / (TPB * 8); ++i) {   // 16 x b128 per lane
        int e = (i * TPB + tid) * 8;
        int c = e >> 9;            // local column 0..127
        int k = e & (NS - 1);      // K row 0..511, multiple of 8
        const half_t* gp = &src[((size_t)(np * PANEL + c)) * NS + k];
        half_t* lp = &lbuf[(k >> 5) * CHUNK_H + c * COL_STR + (k & 31)];
        __builtin_amdgcn_global_load_async_to_lds_b128(
            (AS1 async_v4i*)(AS1 void*)gp,
            (AS3 async_v4i*)(AS3 void*)lp, 0, 0);
    }
#endif
}

// ----------------------------------------------------- per-segment GEMM chain (WMMA)
__global__ __launch_bounds__(TPB, 1)
void dfa_segment_kernel(const int* __restrict__ syms,
                        const half_t* __restrict__ Af16,
                        half_t* __restrict__ bufA,
                        half_t* __restrict__ bufB, int lseg) {
    extern __shared__ half_t lds[];      // 2 * PANEL_H halfs (double-buffered)

    const int p    = blockIdx.x;
    const int tid  = threadIdx.x;
    const int wave = tid >> 5;
    const int lane = tid & 31;
    const int lo   = lane & 15;
    const int hi   = lane >> 4;

    half_t* bA = bufA + (size_t)p * NS * NS;
    half_t* bB = bufB + (size_t)p * NS * NS;
    const int sbase = p * lseg;

    const v8f vzero = {0.f, 0.f, 0.f, 0.f, 0.f, 0.f, 0.f, 0.f};

    for (int t = 1; t < lseg; ++t) {
        const int sym = syms[sbase + t];
        const half_t* __restrict__ A   = Af16 + (size_t)sym * NS * NS;  // row-major
        const half_t* __restrict__ src = (t & 1) ? bA : bB;             // col-major
        half_t* __restrict__       dst = (t & 1) ? bB : bA;             // col-major

#if USE_ASYNC
        stage_panel_async(src, 0, lds, tid);   // prefetch panel 0 of this step
#endif
        for (int np = 0; np < NPANEL; ++np) {
            half_t* cur = lds + (size_t)(np & 1) * PANEL_H;
#if USE_ASYNC
            __builtin_amdgcn_s_wait_asynccnt(0);   // my panel-np async b128s landed
            __syncthreads();                       // everyone's landed; prev panel consumed
            if (np + 1 < NPANEL)                   // overlap next copy with this compute
                stage_panel_async(src, np + 1, lds + (size_t)((np + 1) & 1) * PANEL_H, tid);
#else
            __syncthreads();
            #pragma unroll
            for (int i = 0; i < (PANEL * NS) / (TPB * 8); ++i) {
                int e = (i * TPB + tid) * 8;
                int c = e >> 9;
                int k = e & (NS - 1);
                *(v8h*)&cur[(k >> 5) * CHUNK_H + c * COL_STR + (k & 31)] =
                    *(const v8h*)&src[((size_t)(np * PANEL + c)) * NS + k];
            }
            __syncthreads();
#endif
            // Wave owns output tile rows {wave, wave+16}; each B fragment loaded
            // once from LDS and fed to two WMMAs (halves DS traffic per WMMA).
            v8f acc0[8], acc1[8];
            #pragma unroll
            for (int tn = 0; tn < 8; ++tn) { acc0[tn] = vzero; acc1[tn] = vzero; }

            const half_t* __restrict__ Arow0 = A + (size_t)(wave * 16 + lo) * NS;
            const half_t* __restrict__ Arow1 = Arow0 + (size_t)256 * NS;  // tm = wave+16

            // Single per-lane B base: every fragment below is base + immediate
            // (ks*10240B + tn*1280B + hi*16B (+32B) stays < 64 KB per unroll block).
            const half_t* __restrict__ bwave = &cur[lo * COL_STR + hi * 8];

            #pragma unroll 4
            for (int ks = 0; ks < 16; ++ks) {
                const int kb = ks * 32;
                // A fragments: 16x32 f16; lane lo = row, K chunks {0..7,16..23}/{8..15,24..31}
                v16h a0 = CAT16(*(const v8h*)(Arow0 + kb + hi * 8),
                                *(const v8h*)(Arow0 + kb + 16 + hi * 8));
                v16h a1 = CAT16(*(const v8h*)(Arow1 + kb + hi * 8),
                                *(const v8h*)(Arow1 + kb + 16 + hi * 8));
                #pragma unroll
                for (int tn = 0; tn < 8; ++tn) {
                    // B fragment: 32x16 f16; lane lo = column, K chunks {0..7,16..23}/{8..15,24..31}
                    const half_t* bp = bwave + ks * CHUNK_H + tn * (16 * COL_STR);
                    v16h b = CAT16(*(const v8h*)bp, *(const v8h*)(bp + 16));
                    acc0[tn] = WMMA_F16(a0, b, acc0[tn]);
                    acc1[tn] = WMMA_F16(a1, b, acc1[tn]);
                }
            }

            // Write C tiles to col-major dst: lane lo = col N, rows M = hi*8 + 0..7
            #pragma unroll
            for (int tn = 0; tn < 8; ++tn) {
                v8h o0, o1;
                #pragma unroll
                for (int e = 0; e < 8; ++e) {
                    o0[e] = (half_t)acc0[tn][e];
                    o1[e] = (half_t)acc1[tn][e];
                }
                const int col = np * PANEL + tn * 16 + lo;
                *(v8h*)&dst[(size_t)col * NS + wave * 16 + hi * 8]       = o0;
                *(v8h*)&dst[(size_t)col * NS + wave * 16 + 256 + hi * 8] = o1;
            }
        }
        __threadfence();   // dst (L2) visible to whole workgroup before it becomes src
        __syncthreads();
    }
}

// ------------------------------------------- combine: q = M_{P-1}...M_0 e0 ; dot(q,f)
__global__ __launch_bounds__(NS)
void dfa_combine_kernel(const half_t* __restrict__ bufFinal,
                        const float* __restrict__ f,
                        float* __restrict__ out, int pseg) {
    __shared__ float q[NS];
    __shared__ float red[NS];
    const int m = threadIdx.x;
    q[m] = (m == 0) ? 1.0f : 0.0f;
    __syncthreads();
    for (int p = 0; p < pseg; ++p) {
        const half_t* M = bufFinal + (size_t)p * NS * NS;  // col-major M[n*NS + m]
        float s = 0.f;
        for (int n = 0; n < NS; ++n) s += (float)M[(size_t)n * NS + m] * q[n];
        __syncthreads();
        q[m] = s;
        __syncthreads();
    }
    red[m] = q[m] * f[m];
    __syncthreads();
    for (int w = NS / 2; w > 0; w >>= 1) {
        if (m < w) red[m] += red[m + w];
        __syncthreads();
    }
    if (m == 0) out[0] = red[0];
}

// ---------------------------------------------------------------------------
extern "C" void kernel_launch(void* const* d_in, const int* in_sizes, int n_in,
                              void* d_out, int out_size, void* d_ws, size_t ws_size,
                              hipStream_t stream) {
    const int*   syms  = (const int*)d_in[0];
    const float* delta = (const float*)d_in[1];
    const float* f     = (const float*)d_in[2];
    float*       out   = (float*)d_out;

    const size_t deltaElems = (size_t)NSYMS * NS * NS;           // 16.8M halfs = 32 MB
    const size_t matBytes   = (size_t)NS * NS * sizeof(half_t);  // 512 KB

    // Pick segment count by available scratch: 256 fills all WGPs; fall back if tight.
    int pseg = 256;
    while (pseg > 32 &&
           deltaElems * sizeof(half_t) + 2 * (size_t)pseg * matBytes > ws_size)
        pseg >>= 1;
    const int lseg = SEQLEN / pseg;

    char* ws = (char*)d_ws;
    half_t* Af16 = (half_t*)ws;
    half_t* bufA = (half_t*)(ws + deltaElems * sizeof(half_t));
    half_t* bufB = (half_t*)(ws + deltaElems * sizeof(half_t) + (size_t)pseg * matBytes);

    // Double-buffered K-chunk-major panels: 2 * 160 KB = 320 KB (full WGP LDS).
    const size_t ldsBytes = (size_t)2 * PANEL_H * sizeof(half_t);

    dfa_convert_kernel<<<2048, 256, 0, stream>>>(delta, Af16, (int)deltaElems);
    dfa_init_kernel<<<pseg, 256, 0, stream>>>(syms, Af16, bufA, lseg);
    dfa_segment_kernel<<<pseg, TPB, ldsBytes, stream>>>(syms, Af16, bufA, bufB, lseg);
    // lseg is even -> lseg-1 steps, last (odd t) writes bufB.
    dfa_combine_kernel<<<1, NS, 0, stream>>>(bufB, f, out, pseg);
}
